// SelfAttentionGroundTruth_42623255445965
// MI455X (gfx1250) — matrix operand order repaired and checked by reading, not compiled, not run on hardware
//
#include <hip/hip_runtime.h>
#include <hip/hip_bf16.h>

typedef __attribute__((ext_vector_type(16))) __bf16 v16bf;
typedef __attribute__((ext_vector_type(8)))  __bf16 v8bf;
typedef __attribute__((ext_vector_type(4)))  __bf16 v4bf;
typedef __attribute__((ext_vector_type(8)))  float  v8f;
typedef __attribute__((ext_vector_type(4)))  float  f4;

#define NB   4
#define NH   16
#define SEQ  2048
#define DIM  128
#define QT   64      // query rows per block (4 waves x 16)
#define KT   64      // keys per iteration

static __device__ __forceinline__ v16bf mk16(v8bf lo, v8bf hi) {
  v16bf r;
#pragma unroll
  for (int j = 0; j < 8; ++j) { r[j] = lo[j]; r[j + 8] = hi[j]; }
  return r;
}

static __device__ __forceinline__ v8bf f8_to_bf(const float* __restrict__ p) {
  v8bf r;
#pragma unroll
  for (int j = 0; j < 8; ++j) r[j] = (__bf16)p[j];
  return r;
}

// B-fragment: bf16 p[0..7] and p[16..23] (16B aligned)
static __device__ __forceinline__ v16bf ldB(const __bf16* p) {
  return mk16(*(const v8bf*)p, *(const v8bf*)(p + 16));
}

__global__ __launch_bounds__(128) void
flash_attn_wmma_bf16(const float* __restrict__ Qg, const float* __restrict__ Kg,
                     const float* __restrict__ Vg, const float* __restrict__ Mk,
                     float* __restrict__ Og) {
  __shared__ __align__(16) __bf16 sK[KT][136];        // K tile, row-major [key][d]
  __shared__ __align__(16) __bf16 sVT[DIM][72];       // V tile, transposed [d][key]
  __shared__ __align__(16) __bf16 sP[4][16][72];      // per-wave P scratch [q][key]

  const int qt   = blockIdx.x;          // 0..31
  const int h    = blockIdx.y;
  const int b    = blockIdx.z;
  const int tid  = threadIdx.x;
  const int wave = tid >> 5;
  const int lane = tid & 31;
  const int ln   = lane & 15;
  const int lh   = lane >> 4;

  const int    q0   = qt * QT;
  const size_t head = ((size_t)b * NH + h) * (size_t)SEQ * DIM;
  const float* Qp = Qg + head;
  const float* Kp = Kg + head;
  const float* Vp = Vg + head;
  float*       Op = Og + head;

  const float scale = 0.08838834764831845f;   // 1/sqrt(128)

  // ---- Q A-fragments: wave's 16 rows, 4 chunks of K=32 over D=128 ----
  const int qrow = q0 + wave * 16 + ln;
  v16bf aQ[4];
#pragma unroll
  for (int c = 0; c < 4; ++c) {
    const float* base = Qp + (size_t)qrow * DIM + c * 32 + lh * 8;
    aQ[c] = mk16(f8_to_bf(base), f8_to_bf(base + 16));
  }

  v8f accO[8];
#pragma unroll
  for (int dt = 0; dt < 8; ++dt)
#pragma unroll
    for (int i = 0; i < 8; ++i) accO[dt][i] = 0.0f;

  float mrow[8], lrow[8];
#pragma unroll
  for (int i = 0; i < 8; ++i) { mrow[i] = -__builtin_inff(); lrow[i] = 0.0f; }

  for (int ko = 0; ko < SEQ; ko += KT) {
    __syncthreads();   // protect LDS tiles against prior iteration's readers

    // ---- stage K (row-major bf16) and V (transposed bf16) ----
#pragma unroll
    for (int u = tid; u < KT * (DIM / 4); u += 128) {
      const int r  = u >> 5;          // key row 0..63
      const int c4 = u & 31;          // float4 column
      const f4 kv = *(const f4*)(Kp + (size_t)(ko + r) * DIM + c4 * 4);
      v4bf w;
#pragma unroll
      for (int j = 0; j < 4; ++j) w[j] = (__bf16)kv[j];
      *(v4bf*)&sK[r][c4 * 4] = w;
      const f4 vv = *(const f4*)(Vp + (size_t)(ko + r) * DIM + c4 * 4);
#pragma unroll
      for (int j = 0; j < 4; ++j) sVT[c4 * 4 + j][r] = (__bf16)vv[j];
    }

    // ---- prefetch next iteration's K/V tiles (32KB each = 2x128B lines/thread)
    //      and next mask tile (16KB = 1 line/thread). Issued BEFORE the barrier:
    //      prefetches carry no counters, so they fly while waves sit in
    //      s_barrier_wait, and their address math stays out of the WMMA phase's
    //      register live range. ----
    if (ko + KT < SEQ) {
      const float* nk = Kp + (size_t)(ko + KT) * DIM;
      const float* nv = Vp + (size_t)(ko + KT) * DIM;
#pragma unroll
      for (int j = 0; j < 2; ++j) {
        __builtin_prefetch(nk + (size_t)(tid * 2 + j) * 32, 0, 3);
        __builtin_prefetch(nv + (size_t)(tid * 2 + j) * 32, 0, 3);
      }
      __builtin_prefetch(Mk + (size_t)(q0 + (tid >> 1)) * SEQ + (ko + KT) + (tid & 1) * 32, 0, 3);
    }
    __syncthreads();

    // ---- S = Q x K^T  (4 key sub-tiles x 4 d-chunks of WMMA) ----
    float sv[4][8];
#pragma unroll
    for (int t = 0; t < 4; ++t) {
      v8f acc;
#pragma unroll
      for (int i = 0; i < 8; ++i) acc[i] = 0.0f;
#pragma unroll
      for (int c = 0; c < 4; ++c) {
        v16bf bK = ldB(&sK[t * 16 + ln][c * 32 + lh * 8]);
        acc = __builtin_amdgcn_wmma_f32_16x16x32_bf16(
            false, aQ[c], false, bK, (short)0, acc, false, false);
      }
      const int kcol = ko + t * 16 + ln;
#pragma unroll
      for (int i = 0; i < 8; ++i) {
        const int qr = q0 + wave * 16 + i + 8 * lh;
        sv[t][i] = (acc[i] + Mk[(size_t)qr * SEQ + kcol]) * scale;
      }
    }

    // ---- online softmax (rows live across 16-lane halves) ----
    float rmax[8];
#pragma unroll
    for (int i = 0; i < 8; ++i) {
      float m = sv[0][i];
#pragma unroll
      for (int t = 1; t < 4; ++t) m = fmaxf(m, sv[t][i]);
      rmax[i] = m;
    }
#pragma unroll
    for (int off = 1; off < 16; off <<= 1)
#pragma unroll
      for (int i = 0; i < 8; ++i)
        rmax[i] = fmaxf(rmax[i], __shfl_xor(rmax[i], off, 32));

    float alpha[8];
#pragma unroll
    for (int i = 0; i < 8; ++i) {
      const float mn = fmaxf(mrow[i], rmax[i]);
      alpha[i] = __expf(mrow[i] - mn);
      mrow[i]  = mn;
    }
    float rsum[8];
#pragma unroll
    for (int i = 0; i < 8; ++i) rsum[i] = 0.0f;
#pragma unroll
    for (int t = 0; t < 4; ++t)
#pragma unroll
      for (int i = 0; i < 8; ++i) {
        const float p = __expf(sv[t][i] - mrow[i]);
        sv[t][i] = p;
        rsum[i] += p;
      }
#pragma unroll
    for (int off = 1; off < 16; off <<= 1)
#pragma unroll
      for (int i = 0; i < 8; ++i)
        rsum[i] += __shfl_xor(rsum[i], off, 32);
#pragma unroll
    for (int i = 0; i < 8; ++i) lrow[i] = lrow[i] * alpha[i] + rsum[i];

    // rescale O accumulator
#pragma unroll
    for (int dt = 0; dt < 8; ++dt)
#pragma unroll
      for (int i = 0; i < 8; ++i) accO[dt][i] *= alpha[i];

    // ---- transpose P via per-wave LDS scratch: C-layout -> row-major ----
#pragma unroll
    for (int t = 0; t < 4; ++t)
#pragma unroll
      for (int i = 0; i < 8; ++i)
        sP[wave][i + 8 * lh][t * 16 + ln] = (__bf16)sv[t][i];
    asm volatile("s_wait_dscnt 0" ::: "memory");

    v16bf aP[2];
#pragma unroll
    for (int kk = 0; kk < 2; ++kk) {
      const __bf16* pr = &sP[wave][ln][kk * 32 + lh * 8];
      aP[kk] = mk16(*(const v8bf*)pr, *(const v8bf*)(pr + 16));
    }

    // ---- O += P x V  (8 d-tiles x 2 key-chunks of WMMA) ----
#pragma unroll
    for (int dt = 0; dt < 8; ++dt)
#pragma unroll
      for (int kk = 0; kk < 2; ++kk) {
        v16bf bV = ldB(&sVT[dt * 16 + ln][kk * 32 + lh * 8]);
        accO[dt] = __builtin_amdgcn_wmma_f32_16x16x32_bf16(
            false, aP[kk], false, bV, (short)0, accO[dt], false, false);
      }
  }

  // ---- epilogue: divide by row sum, write fp32 ----
#pragma unroll
  for (int dt = 0; dt < 8; ++dt)
#pragma unroll
    for (int i = 0; i < 8; ++i) {
      const int qr = q0 + wave * 16 + i + 8 * lh;
      Op[(size_t)qr * DIM + dt * 16 + ln] = accO[dt][i] / lrow[i];
    }
}

extern "C" void kernel_launch(void* const* d_in, const int* in_sizes, int n_in,
                              void* d_out, int out_size, void* d_ws, size_t ws_size,
                              hipStream_t stream) {
  (void)in_sizes; (void)n_in; (void)d_ws; (void)ws_size; (void)out_size;
  const float* q  = (const float*)d_in[0];
  const float* k  = (const float*)d_in[1];
  const float* v  = (const float*)d_in[2];
  const float* mk = (const float*)d_in[3];
  float* out = (float*)d_out;

  dim3 grid(SEQ / QT, NH, NB);   // (32, 16, 4)
  dim3 block(128);
  flash_attn_wmma_bf16<<<grid, block, 0, stream>>>(q, k, v, mk, out);
}
